// OutputModule_82738249990283
// MI455X (gfx1250) — matrix operand order repaired and checked by reading, not compiled, not run on hardware
//
#include <hip/hip_runtime.h>
#include <stdint.h>

// ---------------------------------------------------------------------------
// Fused PaiNN-style output head for MI455X (gfx1250), wave32 + WMMA bf16.
//
//   e_feat = ResMLP_e(T) * (rbf(dist) @ e_rbf_w);  e = e_feat @ e_head_w
//   f_feat = ResMLP_f(T) * (rbf(dist) @ f_rbf_w);  f = (f_feat @ f_head_w)*vhat
//   energy[b]    = sum_{i,j} e*mask / 3600
//   forces[j,b,:]= sum_{i}   f*mask / 60
//
// Design: single fused pass over T (268 MB -> ~11.5us HBM floor @23.3TB/s),
// all GEMMs as v_wmma_f32_16x16x32_bf16 (f32 accum), all weights packed once
// into WMMA-fragment order in LDS (256KB + 32KB RBF of the 320KB WGP LDS),
// persistent workgroups, LDS energy accumulation, global f32 atomics for
// forces. Layer loop fully unrolled so per-layer behavior (residual /
// activation) is compile-time; SiLU uses v_rcp_f32 instead of IEEE divide.
// ---------------------------------------------------------------------------

typedef __attribute__((ext_vector_type(16))) __bf16 v16bf;
typedef __attribute__((ext_vector_type(8)))  float  v8f;

#define A_ATOMS 64
#define BATCH   128
#define EMBED   128
#define NG      50
#define R_TOTAL (A_ATOMS*A_ATOMS*BATCH)       // 524288 feature rows
#define WAVES   6
#define TPB     (WAVES*32)                    // 192 threads
#define ROWS_PER_CHUNK (WAVES*16)             // 96 rows per WG iteration
#define NCHUNK  ((R_TOTAL + ROWS_PER_CHUNK - 1)/ROWS_PER_CHUNK)
#define NWG     512

#define RBF_STEP  (12.0f/49.0f)
#define RBF_COEFF (-0.5f/(RBF_STEP*RBF_STEP))

// dynamic LDS layout (bytes)
#define OFF_WFRAG 0            // 8 mats * (4 kt * 8 nt * 512) bf16 = 256 KB
#define OFF_RBF   262144       // 2 mats * (2 kt * 8 nt * 512) bf16 =  32 KB
#define OFF_STAGE 294912       // 6 waves * 16x128 bf16             =  24 KB
#define DYN_LDS   319488

union AFrag { v16bf v; uint32_t u[8]; };

// ISA 16-bit A-matrix 16x32 layout: VGPR v holds K pair; lanes 16-31 offset +8.
__device__ __forceinline__ int a_basek(int v, int h){
  return (v < 4 ? 2*v : 16 + 2*(v-4)) + 8*h;
}
__device__ __forceinline__ uint32_t pack_bf2(float lo, float hi){
  __bf16 a = (__bf16)lo, b = (__bf16)hi;
  return ((uint32_t)__builtin_bit_cast(uint16_t, b) << 16) |
          (uint32_t)__builtin_bit_cast(uint16_t, a);
}
__device__ __forceinline__ v8f wmma_bf16(v16bf a, v16bf b, v8f c){
  // (neg_a, A, neg_b, B, c_mod, C, reuse_a, reuse_b)
  return __builtin_amdgcn_wmma_f32_16x16x32_bf16(false, a, false, b,
                                                 (short)0, c, false, false);
}
// SiLU with hardware reciprocal (v_rcp_f32) -- avoid the IEEE div expansion.
__device__ __forceinline__ float silu_f(float x){
  return x * __builtin_amdgcn_rcpf(1.0f + __expf(-x));
}
__device__ __forceinline__ float gauss_f(float d, int g){
  if (g >= NG) return 0.0f;                    // zero-pad K 50 -> 64
  float t = d - (float)g * RBF_STEP;
  return __expf(RBF_COEFF * t * t);
}

extern "C" __global__ void __launch_bounds__(TPB, 1)
painn_fused(const float* __restrict__ T,     const float* __restrict__ mask,
            const float* __restrict__ dist,  const float* __restrict__ vhat,
            const float* __restrict__ eW_in, const float* __restrict__ eb_in,
            const float* __restrict__ eW_h,  const float* __restrict__ eb_h,
            const float* __restrict__ eW_out,const float* __restrict__ eb_out,
            const float* __restrict__ e_rbf, const float* __restrict__ e_head,
            const float* __restrict__ fW_in, const float* __restrict__ fb_in,
            const float* __restrict__ fW_h,  const float* __restrict__ fb_h,
            const float* __restrict__ fW_out,const float* __restrict__ fb_out,
            const float* __restrict__ f_rbf, const float* __restrict__ f_head,
            float* __restrict__ out)
{
  extern __shared__ char smem[];
  __bf16* wf       = (__bf16*)(smem + OFF_WFRAG);
  __bf16* rf       = (__bf16*)(smem + OFF_RBF);
  __bf16* stageAll = (__bf16*)(smem + OFF_STAGE);
  __shared__ float bias_s[8*EMBED];
  __shared__ float head_s[2*EMBED];
  __shared__ float sE[BATCH];

  const int tid = threadIdx.x;

  // ---- one-time pack: f32 weights -> bf16 WMMA B-fragments in LDS ----
  // B fragment convention (dense 32-K B tile): lane%16 = output column n,
// lanes 0-15 carry K=0..7,16..23, lanes 16-31 carry K=8..15,24..31 of each 32-K tile;
  // 16 bf16 per lane stored contiguously (32B) => two ds_load_b128 at use.
  {
    const float* Wsrc[8] = { eW_in, eW_h, eW_h+EMBED*EMBED, eW_out,
                             fW_in, fW_h, fW_h+EMBED*EMBED, fW_out };
#pragma unroll
    for (int mat = 0; mat < 8; ++mat){
      const float* W = Wsrc[mat];
      for (int idx = tid; idx < 16384; idx += TPB){
        int e = idx & 15, lane = (idx>>4)&31, nt = (idx>>9)&7, kt = idx>>12;
        int K = kt*32 + (lane>>4)*8 + (e < 8 ? e : e + 8);
        int N = nt*16 + (lane&15);
        wf[mat*16384 + idx] = (__bf16)W[K*EMBED + N];
      }
    }
    const float* Rsrc[2] = { e_rbf, f_rbf };
#pragma unroll
    for (int m = 0; m < 2; ++m){
      const float* W = Rsrc[m];
      for (int idx = tid; idx < 8192; idx += TPB){
        int e = idx & 15, lane = (idx>>4)&31, nt = (idx>>9)&7, kt = idx>>12;
        int K = kt*32 + (lane>>4)*8 + (e < 8 ? e : e + 8);
        int N = nt*16 + (lane&15);
        rf[m*8192 + idx] = (__bf16)((K < NG) ? W[K*EMBED + N] : 0.0f);
      }
    }
    const float* Bsrc[8] = { eb_in, eb_h, eb_h+EMBED, eb_out,
                             fb_in, fb_h, fb_h+EMBED, fb_out };
#pragma unroll
    for (int s = 0; s < 8; ++s)
      for (int idx = tid; idx < EMBED; idx += TPB)
        bias_s[s*EMBED + idx] = Bsrc[s][idx];
    for (int idx = tid; idx < EMBED; idx += TPB){
      head_s[idx]        = e_head[idx];
      head_s[EMBED+idx]  = f_head[idx];
    }
    for (int idx = tid; idx < BATCH; idx += TPB) sE[idx] = 0.0f;
  }
  __syncthreads();

  const int wv = tid >> 5, lane = tid & 31;
  const int nl = lane & 15, hh = lane >> 4;
  __bf16*   stage   = stageAll + wv * (16*EMBED);
  uint32_t* stage32 = (uint32_t*)stage;

  // ---- persistent chunk loop: 16 rows per wave per chunk ----
  for (int c = blockIdx.x; c < NCHUNK; c += gridDim.x){
    const int r0 = c*ROWS_PER_CHUNK + wv*16;
    if (r0 >= R_TOTAL) continue;               // wave-uniform, no barriers inside
    const int jj = (r0 >> 7) & (A_ATOMS-1);    // second atom index (forces dim)
    const int b0 = r0 & (BATCH-1);             // batch base (tile stays in-block)

    // prefetch next chunk's tile (gfx1250 global_prefetch_b8)
    {
      size_t rnext = (size_t)(c + gridDim.x)*ROWS_PER_CHUNK + (size_t)wv*16;
      if (rnext < (size_t)R_TOTAL)
        __builtin_prefetch(T + rnext*EMBED + (size_t)lane*16, 0, 1);
    }

    // RBF smeared A-fragments (shared between both MLPs); lane row = nl
    const float d = dist[r0 + nl];
    AFrag ra[2];
#pragma unroll
    for (int kt = 0; kt < 2; ++kt)
#pragma unroll
      for (int v = 0; v < 8; ++v){
        int g = kt*32 + a_basek(v, hh);
        ra[kt].u[v] = pack_bf2(gauss_f(d, g), gauss_f(d, g+1));
      }

#pragma unroll
    for (int mlp = 0; mlp < 2; ++mlp){
      // stage 16x128 f32 tile of T as bf16 (second pass re-reads from cache)
      const float4* src = (const float4*)(T + (size_t)r0*EMBED);
#pragma unroll
      for (int it = 0; it < 16; ++it){
        float4 fv = src[it*32 + lane];
        stage32[(it*32+lane)*2 + 0] = pack_bf2(fv.x, fv.y);
        stage32[(it*32+lane)*2 + 1] = pack_bf2(fv.z, fv.w);
      }

      const int mb = mlp*4;
      v8f outacc[8];
      // Fully unrolled: per-layer residual/activation behavior folds at
      // compile time (no per-nt branches in the hot loop).
#pragma unroll
      for (int layer = 0; layer < 4; ++layer){
        // A fragments from staged activations (loaded fully before overwrite)
        AFrag a[4];
#pragma unroll
        for (int kt = 0; kt < 4; ++kt)
#pragma unroll
          for (int v = 0; v < 8; ++v)
            a[kt].u[v] = *(const uint32_t*)(stage + nl*EMBED + kt*32 + a_basek(v, hh));

        const __bf16* wbase = wf + (mb + layer)*16384;
#pragma unroll
        for (int nt = 0; nt < 8; ++nt){
          v8f cacc = {};
#pragma unroll
          for (int kt = 0; kt < 4; ++kt){
            v16bf bfr = *(const v16bf*)(wbase + (kt*8 + nt)*512 + lane*16);
            cacc = wmma_bf16(a[kt].v, bfr, cacc);
          }
          const int   n  = nt*16 + nl;
          const float bv = bias_s[(mb+layer)*EMBED + n];
          if (layer == 3){
#pragma unroll
            for (int i = 0; i < 8; ++i) cacc[i] += bv;   // out layer: no act
            outacc[nt] = cacc;
          } else {
            // C/D layout: lane holds column n, rows i + 8*hh
#pragma unroll
            for (int i = 0; i < 8; ++i){
              float x = silu_f(cacc[i] + bv);
              if (layer > 0) x += (float)stage[(i + 8*hh)*EMBED + n]; // residual
              stage[(i + 8*hh)*EMBED + n] = (__bf16)x;
            }
          }
        }
      }

      // RBF projection (K=64 in two WMMA steps) + head contraction
      float pr[8] = {0,0,0,0,0,0,0,0};
      const __bf16* rbase = rf + mlp*8192;
#pragma unroll
      for (int nt = 0; nt < 8; ++nt){
        v8f rc = {};
#pragma unroll
        for (int kt = 0; kt < 2; ++kt){
          v16bf bfr = *(const v16bf*)(rbase + (kt*8 + nt)*512 + lane*16);
          rc = wmma_bf16(ra[kt].v, bfr, rc);
        }
        const int   n  = nt*16 + nl;
        const float hw = head_s[mlp*EMBED + n];
#pragma unroll
        for (int i = 0; i < 8; ++i) pr[i] += outacc[nt][i] * rc[i] * hw;
      }
      // reduce over the 16 lanes of each half (columns) -> per-row scalar
#pragma unroll
      for (int m = 1; m < 16; m <<= 1)
#pragma unroll
        for (int i = 0; i < 8; ++i) pr[i] += __shfl_xor(pr[i], m, 16);

      if (nl == 0){
#pragma unroll
        for (int i = 0; i < 8; ++i){
          const int   mrow = i + 8*hh;                 // rows 0-7 / 8-15
          const int   r    = r0 + mrow;
          const float mv   = mask[r];
          if (mlp == 0){
            atomicAdd(&sE[b0 + mrow], pr[i] * mv * (1.0f/3600.0f));
          } else {
            const float  fs    = pr[i] * mv * (1.0f/60.0f);
            const size_t fbase = (size_t)BATCH + ((size_t)jj*BATCH + (b0 + mrow))*3;
            unsafeAtomicAdd(out + fbase + 0, fs * vhat[(size_t)r*3 + 0]);
            unsafeAtomicAdd(out + fbase + 1, fs * vhat[(size_t)r*3 + 1]);
            unsafeAtomicAdd(out + fbase + 2, fs * vhat[(size_t)r*3 + 2]);
          }
        }
      }
    } // mlp
  } // chunk loop

  __syncthreads();
  for (int idx = tid; idx < BATCH; idx += TPB)
    unsafeAtomicAdd(out + idx, sE[idx]);
}

// ---------------------------------------------------------------------------
extern "C" void kernel_launch(void* const* d_in, const int* in_sizes, int n_in,
                              void* d_out, int out_size, void* d_ws, size_t ws_size,
                              hipStream_t stream)
{
  (void)in_sizes; (void)n_in; (void)d_ws; (void)ws_size;
  const float* T      = (const float*)d_in[0];
  const float* mask   = (const float*)d_in[1];
  const float* dist   = (const float*)d_in[2];
  const float* vhat   = (const float*)d_in[3];
  const float* eW_in  = (const float*)d_in[4];
  const float* eb_in  = (const float*)d_in[5];
  const float* eW_h   = (const float*)d_in[6];
  const float* eb_h   = (const float*)d_in[7];
  const float* eW_out = (const float*)d_in[8];
  const float* eb_out = (const float*)d_in[9];
  const float* e_rbf  = (const float*)d_in[10];
  const float* e_head = (const float*)d_in[11];
  const float* fW_in  = (const float*)d_in[12];
  const float* fb_in  = (const float*)d_in[13];
  const float* fW_h   = (const float*)d_in[14];
  const float* fb_h   = (const float*)d_in[15];
  const float* fW_out = (const float*)d_in[16];
  const float* fb_out = (const float*)d_in[17];
  const float* f_rbf  = (const float*)d_in[18];
  const float* f_head = (const float*)d_in[19];
  float* out = (float*)d_out;

  // Outputs are accumulated atomically: zero them every call (capturable).
  hipMemsetAsync(d_out, 0, (size_t)out_size * sizeof(float), stream);

  // Allow >64KB dynamic LDS (320KB WGP LDS on CDNA5). Deterministic each call.
  hipFuncSetAttribute((const void*)painn_fused,
                      hipFuncAttributeMaxDynamicSharedMemorySize, DYN_LDS);

  painn_fused<<<dim3(NWG), dim3(TPB), DYN_LDS, stream>>>(
      T, mask, dist, vhat,
      eW_in, eb_in, eW_h, eb_h, eW_out, eb_out, e_rbf, e_head,
      fW_in, fb_in, fW_h, fb_h, fW_out, fb_out, f_rbf, f_head,
      out);
}
